// MultiHeadLatentAttention_81149112090999
// MI455X (gfx1250) — compile-verified, matrix operand order checked by reading, not run on hardware
//
#include <hip/hip_runtime.h>

// ---------------------------------------------------------------------------
// MLA (Multi-Head Latent Attention) for gfx1250 / MI455X, wave32 + WMMA bf16.
// Outputs (concatenated in d_out): output [B,T,2048] fp32, p_attn [B,H,T,T] fp32
// ---------------------------------------------------------------------------

#define HH   16
#define TT   2048
#define BB   2
#define DM   2048
#define DC   512
#define DR   64
#define DH   128
#define DQK  192

typedef __attribute__((ext_vector_type(16))) __bf16 v16bf;
typedef __attribute__((ext_vector_type(8)))  __bf16 v8bf;
typedef __attribute__((ext_vector_type(2)))  __bf16 v2bf;
typedef __attribute__((ext_vector_type(8)))  float  v8f;
typedef __attribute__((ext_vector_type(4)))  float  v4f;
typedef __attribute__((ext_vector_type(2)))  float  v2f;

__device__ __forceinline__ v8f wmma_bf16(v16bf a, v16bf b, v8f c) {
    // D = A(16x32 bf16) x B(32x16 bf16) + C(16x16 f32)
    return __builtin_amdgcn_wmma_f32_16x16x32_bf16(
        false, a, false, b, (short)0, c, false, false);
}

// A/B fragment from bf16 memory (row of >=32 contiguous bf16 at `base`).
// Per ISA 7.12.2: lane hi=0 -> K = {0..7, 16..23}; hi=1 -> K = {8..15, 24..31}.
__device__ __forceinline__ v16bf load_frag_bf(const __bf16* base, int hi) {
    union { v16bf v; v8bf h[2]; } u;
    u.h[0] = *(const v8bf*)(base + hi * 8);
    u.h[1] = *(const v8bf*)(base + 16 + hi * 8);
    return u.v;
}

// Same fragment but sourced from fp32 memory, converted to bf16 in registers.
__device__ __forceinline__ v16bf load_frag_f32(const float* base, int hi) {
    const v4f* p0 = (const v4f*)(base + hi * 8);
    const v4f* p1 = (const v4f*)(base + 16 + hi * 8);
    v4f x0 = p0[0], x1 = p0[1], x2 = p1[0], x3 = p1[1];
    v16bf r;
#pragma unroll
    for (int i = 0; i < 4; ++i) {
        r[i]      = (__bf16)x0[i];
        r[4 + i]  = (__bf16)x1[i];
        r[8 + i]  = (__bf16)x2[i];
        r[12 + i] = (__bf16)x3[i];
    }
    return r;
}

__device__ __forceinline__ v2bf cvt2(float a, float b) {
    v2bf r; r[0] = (__bf16)a; r[1] = (__bf16)b; return r;
}

// ---------------------------------------------------------------------------
// Generic row-major GEMM: C[MxN] = A[MxK] * B[KxN], fp32 in, fp32 out, bf16
// WMMA math. All dims compile-time so row strides fold into store offsets.
//   Block tile BM x BN, wave grid WR x WC (THREADS = WR*WC*32),
//   wave tile (BM/WR) x (BN/WC), (BM/WR/16) x (BN/WC/16) WMMA accumulators,
//   K-step 32 staged through LDS with paired fp32->bf16 conversion.
// ---------------------------------------------------------------------------
template <int M, int N, int K, int BM, int BN, int WR, int WC>
__global__ __launch_bounds__(WR * WC * 32)
void gemm_bf16_kernel(const float* __restrict__ A, const float* __restrict__ B,
                      float* __restrict__ C) {
    constexpr int THREADS = WR * WC * 32;
    constexpr int WTM = BM / WR;          // wave tile rows
    constexpr int WTN = BN / WC;          // wave tile cols
    constexpr int MF  = WTM / 16;         // m fragments per wave
    constexpr int NF  = WTN / 16;         // n fragments per wave

    __shared__ __bf16 sA[BM * 32];        // [row][k]
    __shared__ __bf16 sB[BN * 32];        // transposed: [n][k]

    const int tid  = threadIdx.x;
    const int lane = tid & 31;
    const int wave = tid >> 5;
    const int wr   = wave / WC;
    const int wc   = wave % WC;
    const int r16  = lane & 15;
    const int hi   = lane >> 4;

    const int row0 = blockIdx.y * BM;
    const int col0 = blockIdx.x * BN;

    v8f acc[MF][NF] = {};

    for (int k0 = 0; k0 < K; k0 += 32) {
        // A tile: pairs along k (contiguous float2 load, packed b32 LDS store)
#pragma unroll
        for (int i = 0; i < (BM * 16) / THREADS; ++i) {
            int p   = tid + THREADS * i;          // pair index
            int r   = p >> 4;
            int c   = (p & 15) * 2;
            v2f a   = *(const v2f*)&A[(size_t)(row0 + r) * K + (k0 + c)];
            *(v2bf*)&sA[r * 32 + c] = cvt2(a[0], a[1]);
        }
        // B tile transposed: pair of k rows -> packed b32 LDS store
#pragma unroll
        for (int i = 0; i < (BN * 16) / THREADS; ++i) {
            int p = tid + THREADS * i;
            int n = p % BN;
            int k = (p / BN) * 2;
            float b0 = B[(size_t)(k0 + k) * N + (col0 + n)];
            float b1 = B[(size_t)(k0 + k + 1) * N + (col0 + n)];
            *(v2bf*)&sB[n * 32 + k] = cvt2(b0, b1);
        }
        __syncthreads();

        v16bf af[MF], bf[NF];
#pragma unroll
        for (int mf = 0; mf < MF; ++mf)
            af[mf] = load_frag_bf(&sA[(wr * WTM + mf * 16 + r16) * 32], hi);
#pragma unroll
        for (int nf = 0; nf < NF; ++nf)
            bf[nf] = load_frag_bf(&sB[(wc * WTN + nf * 16 + r16) * 32], hi);
#pragma unroll
        for (int mf = 0; mf < MF; ++mf)
#pragma unroll
            for (int nf = 0; nf < NF; ++nf)
                acc[mf][nf] = wmma_bf16(af[mf], bf[nf], acc[mf][nf]);
        __syncthreads();
    }

    // C/D layout: VGPR j -> M = 8*hi + j ; N = lane&15
#pragma unroll
    for (int mf = 0; mf < MF; ++mf)
#pragma unroll
        for (int nf = 0; nf < NF; ++nf) {
            int n = col0 + wc * WTN + nf * 16 + r16;
            int rbase = row0 + wr * WTM + mf * 16 + hi * 8;
            float* cp = C + (size_t)rbase * N + n;
#pragma unroll
            for (int j = 0; j < 8; ++j)
                cp[(size_t)j * N] = acc[mf][nf][j];
        }
}

// ---------------------------------------------------------------------------
// Pack Q or K into contiguous (b,h,t,192) bf16 rows, fusing RoPE on the last
// 64 features. ropeHeadStride = 64 for Q (per-head rope), 0 for K (broadcast).
// One thread per feature pair (paired b32 store). total = B*H*T*96.
// ---------------------------------------------------------------------------
__global__ __launch_bounds__(256)
void pack_rope_kernel(const float* __restrict__ nope, const float* __restrict__ rope,
                      __bf16* __restrict__ pack, int ropeRowStride, int ropeHeadStride,
                      int total) {
    int idx = blockIdx.x * blockDim.x + threadIdx.x;
    if (idx >= total) return;
    int p = idx % 96;
    int t = (idx / 96) % TT;
    int h = (idx / (96 * TT)) % HH;
    int b = idx / (96 * TT * HH);

    size_t inRow   = (size_t)b * TT + t;
    size_t outBase = ((size_t)(b * HH + h) * TT + t) * DQK;

    if (p < 64) {
        int d = p * 2;
        v2f x = *(const v2f*)&nope[inRow * DM + h * DH + d];
        *(v2bf*)&pack[outBase + d] = cvt2(x[0], x[1]);
    } else {
        int i = p - 64;                       // rope pair index 0..31
        const float* rp = rope + inRow * (size_t)ropeRowStride + h * ropeHeadStride;
        float x1 = rp[2 * i], x2 = rp[2 * i + 1];
        // inv_freq = 10000^(-2i/64) = exp(-(i/32)*ln(10000))
        float invf = __expf(-((float)i / 32.0f) * 9.210340372f);
        float ang  = (float)t * invf;
        float c = cosf(ang), s = sinf(ang);
        *(v2bf*)&pack[outBase + DH + 2 * i] = cvt2(x1 * c - x2 * s, x1 * s + x2 * c);
    }
}

// ---------------------------------------------------------------------------
// Pack V transposed: vT[(b,h,d,t)] = v_flat[(b,t), h*128+d], bf16, two t per
// thread (paired b32 store). Reads are strided but L2-resident (32 MB).
// ---------------------------------------------------------------------------
__global__ __launch_bounds__(256)
void pack_vT_kernel(const float* __restrict__ vflat, __bf16* __restrict__ vT, int total) {
    int idx = blockIdx.x * blockDim.x + threadIdx.x;   // total = B*H*T*DH/2
    if (idx >= total) return;
    int t  = (idx % (TT / 2)) * 2;
    int d  = (idx / (TT / 2)) % DH;
    int bh = idx / ((TT / 2) * DH);
    int b = bh >> 4, h = bh & 15;
    float a0 = vflat[((size_t)b * TT + t) * DM + h * DH + d];
    float a1 = vflat[((size_t)b * TT + t + 1) * DM + h * DH + d];
    *(v2bf*)&vT[((size_t)bh * DH + d) * TT + t] = cvt2(a0, a1);
}

// ---------------------------------------------------------------------------
// Scores: S[bh] = (Q[bh] @ K[bh]^T) / sqrt(192). Q,K are packed (bh,t,192) bf16
// so both A and B fragments are contiguous-in-K global b128 loads (L2-resident).
// Block tile 64x64, 4 waves, K = 192 in 6 WMMA steps.
// ---------------------------------------------------------------------------
__global__ __launch_bounds__(128)
void scores_kernel(const __bf16* __restrict__ qp, const __bf16* __restrict__ kp,
                   float* __restrict__ pattn) {
    const int tid  = threadIdx.x;
    const int lane = tid & 31;
    const int wave = tid >> 5;
    const int wr = wave >> 1, wc = wave & 1;
    const int r16 = lane & 15, hi = lane >> 4;
    const int bh   = blockIdx.z;
    const int row0 = blockIdx.y * 64;
    const int col0 = blockIdx.x * 64;

    const __bf16* qbase = qp + (size_t)bh * TT * DQK;
    const __bf16* kbase = kp + (size_t)bh * TT * DQK;

    v8f acc[2][2] = {};
#pragma unroll
    for (int k0 = 0; k0 < DQK; k0 += 32) {
        v16bf af[2], bf[2];
#pragma unroll
        for (int wm = 0; wm < 2; ++wm)
            af[wm] = load_frag_bf(qbase + (size_t)(row0 + wr * 32 + wm * 16 + r16) * DQK + k0, hi);
#pragma unroll
        for (int wn = 0; wn < 2; ++wn)
            bf[wn] = load_frag_bf(kbase + (size_t)(col0 + wc * 32 + wn * 16 + r16) * DQK + k0, hi);
#pragma unroll
        for (int wm = 0; wm < 2; ++wm)
#pragma unroll
            for (int wn = 0; wn < 2; ++wn)
                acc[wm][wn] = wmma_bf16(af[wm], bf[wn], acc[wm][wn]);
    }

    const float scale = 0.07216878364870323f;   // 1/sqrt(192)
    float* out = pattn + (size_t)bh * TT * TT;
#pragma unroll
    for (int wm = 0; wm < 2; ++wm)
#pragma unroll
        for (int wn = 0; wn < 2; ++wn) {
            int n = col0 + wc * 32 + wn * 16 + r16;
            int rbase = row0 + wr * 32 + wm * 16 + hi * 8;
            float* op = out + (size_t)rbase * TT + n;
#pragma unroll
            for (int j = 0; j < 8; ++j)
                op[(size_t)j * TT] = acc[wm][wn][j] * scale;
        }
}

// ---------------------------------------------------------------------------
// Row softmax in place; one 256-thread block per row of 2048.
// ---------------------------------------------------------------------------
__global__ __launch_bounds__(256)
void softmax_kernel(float* __restrict__ p) {
    __shared__ float red[256];
    const int tid = threadIdx.x;
    float* x = p + (size_t)blockIdx.x * TT;

    float v[8];
    float mx = -3.402823466e38f;
#pragma unroll
    for (int i = 0; i < 8; ++i) { v[i] = x[tid + 256 * i]; mx = fmaxf(mx, v[i]); }
    red[tid] = mx; __syncthreads();
    for (int s = 128; s > 0; s >>= 1) {
        if (tid < s) red[tid] = fmaxf(red[tid], red[tid + s]);
        __syncthreads();
    }
    mx = red[0]; __syncthreads();

    float sum = 0.f;
#pragma unroll
    for (int i = 0; i < 8; ++i) { v[i] = __expf(v[i] - mx); sum += v[i]; }
    red[tid] = sum; __syncthreads();
    for (int s = 128; s > 0; s >>= 1) {
        if (tid < s) red[tid] += red[tid + s];
        __syncthreads();
    }
    float inv = 1.0f / red[0];
#pragma unroll
    for (int i = 0; i < 8; ++i) x[tid + 256 * i] = v[i] * inv;
}

// ---------------------------------------------------------------------------
// x[bh] = P[bh] @ V[bh]. P is fp32 in d_out (probabilities), converted to bf16
// in registers; V is the transposed bf16 pack so B fragments are contiguous.
// Output scattered into x_flat[(b,t), h*128+d] (row-major for final W_O GEMM).
// ---------------------------------------------------------------------------
__global__ __launch_bounds__(128)
void pv_kernel(const float* __restrict__ pattn, const __bf16* __restrict__ vT,
               float* __restrict__ xf) {
    const int tid  = threadIdx.x;
    const int lane = tid & 31;
    const int wave = tid >> 5;
    const int wr = wave >> 1, wc = wave & 1;
    const int r16 = lane & 15, hi = lane >> 4;
    const int bh = blockIdx.z;
    const int b = bh >> 4, h = bh & 15;
    const int row0 = blockIdx.y * 64;
    const int col0 = blockIdx.x * 64;   // 0 or 64 (D_HEAD = 128)

    const float*  P = pattn + (size_t)bh * TT * TT;
    const __bf16* V = vT + (size_t)bh * DH * TT;

    v8f acc[2][2] = {};
    for (int k0 = 0; k0 < TT; k0 += 32) {
        v16bf af[2], bf[2];
#pragma unroll
        for (int wm = 0; wm < 2; ++wm)
            af[wm] = load_frag_f32(P + (size_t)(row0 + wr * 32 + wm * 16 + r16) * TT + k0, hi);
#pragma unroll
        for (int wn = 0; wn < 2; ++wn)
            bf[wn] = load_frag_bf(V + (size_t)(col0 + wc * 32 + wn * 16 + r16) * TT + k0, hi);
#pragma unroll
        for (int wm = 0; wm < 2; ++wm)
#pragma unroll
            for (int wn = 0; wn < 2; ++wn)
                acc[wm][wn] = wmma_bf16(af[wm], bf[wn], acc[wm][wn]);
    }

#pragma unroll
    for (int wm = 0; wm < 2; ++wm)
#pragma unroll
        for (int wn = 0; wn < 2; ++wn) {
            int n = col0 + wc * 32 + wn * 16 + r16;    // 0..127
            int rbase = row0 + wr * 32 + wm * 16 + hi * 8;
            float* op = xf + (size_t)(b * TT + rbase) * DM + h * DH + n;
#pragma unroll
            for (int j = 0; j < 8; ++j)
                op[(size_t)j * DM] = acc[wm][wn][j];
        }
}

// ---------------------------------------------------------------------------
// Launch
// ---------------------------------------------------------------------------
extern "C" void kernel_launch(void* const* d_in, const int* in_sizes, int n_in,
                              void* d_out, int out_size, void* d_ws, size_t ws_size,
                              hipStream_t stream) {
    (void)in_sizes; (void)n_in; (void)out_size; (void)ws_size;

    const float* query = (const float*)d_in[0];
    const float* key   = (const float*)d_in[1];
    /* value (d_in[2]) is unused by the reference: V = c_kv @ W_UV */
    const float* W_DKV = (const float*)d_in[3];
    const float* W_UK  = (const float*)d_in[4];
    const float* W_UV  = (const float*)d_in[5];
    const float* W_Q   = (const float*)d_in[6];
    const float* W_KR  = (const float*)d_in[7];
    const float* W_QR  = (const float*)d_in[8];
    const float* W_O   = (const float*)d_in[9];

    float* out   = (float*)d_out;                          // [B,T,DM]
    float* pattn = out + (size_t)BB * TT * DM;             // [B,H,T,T]

    constexpr int BT = BB * TT;                            // 4096 rows

    // Workspace carve-up (fp32 regions first, then bf16 packs). ~185 MB.
    float* cKV = (float*)d_ws;                             //  8 MB
    float* qN  = cKV + (size_t)BT * DC;                    // 32 MB
    float* qR  = qN  + (size_t)BT * DM;                    // 16 MB
    float* kR  = qR  + (size_t)BT * HH * DR;               //  1 MB
    float* kN  = kR  + (size_t)BT * DR;                    // 32 MB
    float* vF  = kN  + (size_t)BT * DM;                    // 32 MB
    __bf16* qP = (__bf16*)(vF + (size_t)BT * DM);          // 24 MB
    __bf16* kP = qP + (size_t)BB * HH * TT * DQK;          // 24 MB
    __bf16* vT = kP + (size_t)BB * HH * TT * DQK;          // 16 MB
    float* xF  = qN;   // alias: q_nope is dead once packs are done

    dim3 blk128(128), blk256(256);

    // --- Projections (bf16 WMMA GEMMs, 128x128 block / 64x32 wave tiles) ---
    gemm_bf16_kernel<BT, DC, DM, 128, 128, 2, 4>
        <<<dim3(DC / 128, BT / 128), 256, 0, stream>>>(key,   W_DKV, cKV);
    gemm_bf16_kernel<BT, DM, DC, 128, 128, 2, 4>
        <<<dim3(DM / 128, BT / 128), 256, 0, stream>>>(cKV,   W_UK,  kN);
    gemm_bf16_kernel<BT, DM, DC, 128, 128, 2, 4>
        <<<dim3(DM / 128, BT / 128), 256, 0, stream>>>(cKV,   W_UV,  vF);
    gemm_bf16_kernel<BT, DM, DM, 128, 128, 2, 4>
        <<<dim3(DM / 128, BT / 128), 256, 0, stream>>>(query, W_Q,   qN);
    gemm_bf16_kernel<BT, DR, DM, 64, 64, 2, 2>
        <<<dim3(DR / 64, BT / 64), 128, 0, stream>>>(key,   W_KR,  kR);
    gemm_bf16_kernel<BT, HH * DR, DM, 128, 128, 2, 4>
        <<<dim3((HH * DR) / 128, BT / 128), 256, 0, stream>>>(query, W_QR, qR);

    // --- RoPE + pack Q/K (bf16, contiguous 192-wide rows), pack V^T ---
    {
        int total = BB * HH * TT * 96;
        pack_rope_kernel<<<(total + 255) / 256, blk256, 0, stream>>>(qN, qR, qP, HH * DR, DR, total);
        pack_rope_kernel<<<(total + 255) / 256, blk256, 0, stream>>>(kN, kR, kP, DR, 0, total);
        int vtot = BB * HH * TT * DH / 2;
        pack_vT_kernel<<<(vtot + 255) / 256, blk256, 0, stream>>>(vF, vT, vtot);
    }

    // --- Attention ---
    scores_kernel<<<dim3(TT / 64, TT / 64, BB * HH), blk128, 0, stream>>>(qP, kP, pattn);
    softmax_kernel<<<dim3(BB * HH * TT), blk256, 0, stream>>>(pattn);
    pv_kernel<<<dim3(DH / 64, TT / 64, BB * HH), blk128, 0, stream>>>(pattn, vT, xF);

    // --- Output projection ---
    gemm_bf16_kernel<BT, DM, DM, 128, 128, 2, 4>
        <<<dim3(DM / 128, BT / 128), 256, 0, stream>>>(xF, W_O, out);
}
